// STGAT_86741159510109
// MI455X (gfx1250) — matrix operand order repaired and checked by reading, not compiled, not run on hardware
//
#include <hip/hip_runtime.h>
#include <hip/hip_bf16.h>

// ---------------- problem constants (from reference setup_inputs) -----------
#define BB   4
#define TT   12
#define NN   512
#define FF   32
#define HD   64
#define KH   4
#define CC_  8
#define MM   (BB * TT)          // 48
#define R1   (MM * NN)          // 24576 node-rows
#define BN   (BB * NN)          // 2048 sequences
#define CHUNK 128               // k-rows of Wh staged per TDM transfer
#define NCH  (NN / CHUNK)       // 4 chunks

typedef __attribute__((ext_vector_type(16))) _Float16 v16h;
typedef __attribute__((ext_vector_type(8)))  float    v8f;
typedef __attribute__((ext_vector_type(4)))  unsigned int u32x4;
typedef __attribute__((ext_vector_type(8)))  int      i32x8;
typedef __attribute__((ext_vector_type(4)))  int      i32x4;

#if defined(__has_builtin)
#if __has_builtin(__builtin_amdgcn_tensor_load_to_lds)
#define HAVE_TDM 1
#endif
#endif

// ---- WMMA operand loaders (CDNA5 ISA 7.12.2 wave32 layouts), f32 -> f16 ----
// A: 16x32 (MxK), row-major fp32 source with leading dim lda.
__device__ __forceinline__ v16h load_a16x32(const float* __restrict__ src, int lda, int lane) {
  int row = lane & 15, hi = lane >> 4;
  const float* p = src + row * lda;
  v16h a;
#pragma unroll
  for (int e = 0; e < 16; ++e) {
    int k = (e & 7) + (hi << 3) + ((e >> 3) << 4);   // K index per ISA layout
    a[e] = (_Float16)p[k];
  }
  return a;
}
// B: 32x16 (KxN), fp32 source addressed as src[k*sk + col*sc] (strided => free transpose)
__device__ __forceinline__ v16h load_b32x16(const float* __restrict__ src, int sk, int sc, int lane) {
  int col = lane & 15, hi = lane >> 4;
  v16h b;
#pragma unroll
  for (int e = 0; e < 16; ++e) {
    int k = (e & 7) + (hi << 3) + ((e >> 3) << 4);
    b[e] = (_Float16)src[k * sk + col * sc];
  }
  return b;
}

#ifdef HAVE_TDM
// ---- Tensor Data Mover: 2D tile (rows x cols f32) global -> LDS ------------
// D# layout per CDNA5 ISA 8.3/8.4: group0 {count, lds_addr, global_addr, type=2},
// group1 {data_size=4B, tensor_dim0/1, tile_dim0/1, tensor_dim0_stride}.
__device__ __forceinline__ unsigned lds_off(const void* p) {
  return (unsigned)(unsigned long long)p;    // LDS aperture: addr[31:0] = LDS offset
}
__device__ __forceinline__ void tdm_load_2d(const float* gaddr, unsigned ldsa,
                                            unsigned rows, unsigned cols,
                                            unsigned strideElems) {
  unsigned long long ga = (unsigned long long)gaddr;
  u32x4 g0;
  g0[0] = 1u;                                         // count=1 valid descriptor
  g0[1] = ldsa;                                       // lds_addr (bytes)
  g0[2] = (unsigned)(ga & 0xFFFFFFFFu);               // global_addr[31:0]
  g0[3] = (unsigned)((ga >> 32) & 0x01FFFFFFu) | (2u << 30);  // addr[56:32] | type=2
  i32x8 g1;
  g1[0] = (int)(2u << 16);                            // data_size=2 -> 4 bytes; mask=0
  g1[1] = (int)((cols & 0xFFFFu) << 16);              // tensor_dim0[15:0] @ bits 63:48
  g1[2] = (int)(((cols >> 16) & 0xFFFFu) | ((rows & 0xFFFFu) << 16)); // dim0 hi | dim1 lo
  g1[3] = (int)(((rows >> 16) & 0xFFFFu) | ((cols & 0xFFFFu) << 16)); // dim1 hi | tile_dim0
  g1[4] = (int)(rows & 0xFFFFu);                      // tile_dim1 (tile_dim2=0)
  g1[5] = (int)strideElems;                           // tensor_dim0_stride[31:0]
  g1[6] = 0;
  g1[7] = 0;
  i32x4 z4 = {0, 0, 0, 0};
#if __clang_major__ >= 23
  i32x8 z8 = {0, 0, 0, 0, 0, 0, 0, 0};
  __builtin_amdgcn_tensor_load_to_lds(g0, g1, z4, z4, z8, 0);
#else
  __builtin_amdgcn_tensor_load_to_lds(g0, g1, z4, z4, 0);
#endif
}
#endif

// ---------------- generic WMMA GEMM: C[z] = A * B[z] (+bias1+bias2) ---------
// grid.x = rows/16, grid.z = batch of B/C; block = 128 (4 waves); waves split col-tiles.
__global__ __launch_bounds__(128)
void gemm_f16wmma(const float* __restrict__ A, int lda,
                  const float* __restrict__ B, int sk, int sc, int kdim,
                  float* __restrict__ C, int ldc, int ncols,
                  const float* __restrict__ bias1, const float* __restrict__ bias2,
                  long bStrideZ, long cStrideZ) {
  int lane = threadIdx.x & 31, wave = threadIdx.x >> 5;
  const float* Bp = B + (long)blockIdx.z * bStrideZ;
  float*       Cp = C + (long)blockIdx.z * cStrideZ;
  long row0 = (long)blockIdx.x * 16;
  int ntiles = ncols >> 4;
  for (int ct = wave; ct < ntiles; ct += 4) {
    int col0 = ct << 4;
    v8f acc = {};
    for (int kk = 0; kk < kdim; kk += 32) {
      if (kk + 32 < kdim) {   // prefetch next k-panels -> global_prefetch_b8
        __builtin_prefetch((const void*)(A + row0 * lda + kk + 32), 0, 1);
        __builtin_prefetch((const void*)(Bp + (long)(kk + 32) * sk + (long)col0 * sc), 0, 1);
      }
      v16h a = load_a16x32(A + row0 * lda + kk, lda, lane);
      v16h b = load_b32x16(Bp + (long)kk * sk + (long)col0 * sc, sk, sc, lane);
      acc = __builtin_amdgcn_wmma_f32_16x16x32_f16(false, a, false, b, (short)0, acc, false, false);
    }
    int col = lane & 15, hi = lane >> 4;
    float bs = 0.f;
    if (bias1) bs += bias1[col0 + col];
    if (bias2) bs += bias2[col0 + col];
#pragma unroll
    for (int v = 0; v < 8; ++v)
      Cp[(row0 + v + (hi << 3)) * ldc + col0 + col] = acc[v] + bs;
  }
}

// ---------------- f1/f2 = Wh . a[:H], Wh . a[H:] per node-row ---------------
__global__ void f1f2_kernel(const float* __restrict__ X, const float* __restrict__ a,
                            float* __restrict__ f1, float* __restrict__ f2, long rowsPerZ) {
  long r = (long)blockIdx.z * rowsPerZ + (long)blockIdx.x * blockDim.x + threadIdx.x;
  const float* av = a + (long)blockIdx.z * (2 * HD);
  const float* xr = X + r * HD;
  float s1 = 0.f, s2 = 0.f;
#pragma unroll 8
  for (int h = 0; h < HD; ++h) { float v = xr[h]; s1 += v * av[h]; s2 += v * av[HD + h]; }
  f1[r] = s1; f2[r] = s2;
}

// ---------------- GAT attention + softmax + attn@Wh + ELU ------------------
// grid = (N/16, M, K). mode 0: write hcat[(m*N+i)*256 + k*64 + h]
//                      mode 1: write xseq[((b*N+i)*T + t)*64 + h], m = b*T+t
// Wh is staged 128x64 at a time into LDS by the Tensor Data Mover (wave 0),
// double-buffered and pipelined against the WMMA consumption of the prior chunk.
__global__ __launch_bounds__(128)
void attn_kernel(const float* __restrict__ Wh, const float* __restrict__ f1,
                 const float* __restrict__ f2, const int* __restrict__ adj,
                 float* __restrict__ out, int mode) {
  __shared__ float attn[16 * NN];            // 32 KB logits/probs
  __shared__ float sWh[2][CHUNK * HD];       // 2 x 32 KB staged B operand
  __shared__ float red[16 * 8];
  int k = blockIdx.z, m = blockIdx.y, i0 = blockIdx.x * 16;
  long base = ((long)k * MM + m) * NN;       // node-row base for this (k,m)
  const float* Whm = Wh + base * HD;
  int tid = threadIdx.x;
  bool tdmWave = (tid < 32);                 // uniform per-wave branch (EXEC stays full)

#ifdef HAVE_TDM
  if (tdmWave)                               // kick chunk 0 DMA; hides behind softmax
    tdm_load_2d(Whm, lds_off(&sWh[0][0]), CHUNK, HD, HD);
#endif

  // 1) logits: leakyrelu(f1_i + f2_j), masked
  for (int idx = tid; idx < 16 * NN; idx += 128) {
    int r = idx >> 9, j = idx & (NN - 1);
    float e = f1[base + i0 + r] + f2[base + j];
    e = (e > 0.f) ? e : 0.2f * e;
    if (adj[(i0 + r) * NN + j] == 0) e = -9e15f;
    attn[idx] = e;
  }
  __syncthreads();

  // 2) row softmax: 8 threads per row
  int r = tid >> 3, s = tid & 7;
  float mx = -3.4e38f;
  for (int j = s; j < NN; j += 8) mx = fmaxf(mx, attn[r * NN + j]);
  red[r * 8 + s] = mx;
  __syncthreads();
  float rowmax = red[r * 8];
#pragma unroll
  for (int q = 1; q < 8; ++q) rowmax = fmaxf(rowmax, red[r * 8 + q]);
  __syncthreads();
  float sum = 0.f;
  for (int j = s; j < NN; j += 8) {
    float v = __expf(attn[r * NN + j] - rowmax);
    attn[r * NN + j] = v; sum += v;
  }
  red[r * 8 + s] = sum;
  __syncthreads();
  float tot = 0.f;
#pragma unroll
  for (int q = 0; q < 8; ++q) tot += red[r * 8 + q];
  float inv = 1.f / tot;
  for (int j = s; j < NN; j += 8) attn[r * NN + j] *= inv;
  __syncthreads();

  // 3) O(16x64) = attn(16x512) @ Wh(512x64) via WMMA over TDM-staged chunks
  int lane = tid & 31, wave = tid >> 5;
  int col0 = wave << 4;
  v8f acc = {};
  for (int cb = 0; cb < NCH; ++cb) {
#ifdef HAVE_TDM
    if (tdmWave) {
      if (cb + 1 < NCH) {                    // issue next chunk, wait for current
        tdm_load_2d(Whm + (long)(cb + 1) * CHUNK * HD,
                    lds_off(&sWh[(cb + 1) & 1][0]), CHUNK, HD, HD);
        __builtin_amdgcn_s_wait_tensorcnt(1);
      } else {
        __builtin_amdgcn_s_wait_tensorcnt(0);
      }
    }
#else
    for (int idx = tid; idx < CHUNK * HD; idx += 128)
      sWh[cb & 1][idx] = Whm[(long)cb * CHUNK * HD + idx];
#endif
    __syncthreads();                          // staged chunk visible to all waves
    const float* Bsrc = &sWh[cb & 1][0];
    for (int kk = 0; kk < CHUNK; kk += 32) {
      v16h a = load_a16x32(&attn[cb * CHUNK + kk], NN, lane);
      v16h b = load_b32x16(Bsrc + kk * HD + col0, HD, 1, lane);
      acc = __builtin_amdgcn_wmma_f32_16x16x32_f16(false, a, false, b, (short)0, acc, false, false);
    }
    __syncthreads();                          // done reading buffer before reuse
  }

  // 4) ELU + scatter to layout
  int col = lane & 15, hi = lane >> 4;
#pragma unroll
  for (int v = 0; v < 8; ++v) {
    int i = i0 + v + (hi << 3);
    int h = col0 + col;
    float val = acc[v];
    val = (val > 0.f) ? val : (__expf(val) - 1.f);
    if (mode == 0) {
      out[((long)m * NN + i) * (KH * HD) + k * HD + h] = val;
    } else {
      int b = m / TT, t = m % TT;
      out[(((long)b * NN + i) * TT + t) * HD + h] = val;
    }
  }
}

// ---------------- LSTM pointwise gates (pytorch i,f,g,o order) --------------
__global__ void lstm_gate(const float* __restrict__ pre, const float* __restrict__ g,
                          float* __restrict__ hh, float* __restrict__ cc, int t) {
  int idx = blockIdx.x * blockDim.x + threadIdx.x;   // BN*HD threads
  int rr = idx >> 6, h = idx & 63;
  const float* pr = pre + ((long)rr * TT + t) * (4 * HD);
  const float* gr = g + (long)rr * (4 * HD);
  float i = pr[h]            + gr[h];
  float f = pr[HD + h]       + gr[HD + h];
  float gg = pr[2 * HD + h]  + gr[2 * HD + h];
  float o = pr[3 * HD + h]   + gr[3 * HD + h];
  i = 1.f / (1.f + __expf(-i));
  f = 1.f / (1.f + __expf(-f));
  o = 1.f / (1.f + __expf(-o));
  gg = tanhf(gg);
  float c = f * cc[idx] + i * gg;
  cc[idx] = c;
  hh[idx] = o * tanhf(c);
}

__global__ void final_proj(const float* __restrict__ hh, const float* __restrict__ W_lin,
                           const float* __restrict__ b_lin, float* __restrict__ out) {
  int idx = blockIdx.x * blockDim.x + threadIdx.x;   // BN*C threads
  int rr = idx >> 3, c = idx & 7;
  const float* w = W_lin + c * HD;
  const float* x = hh + (long)rr * HD;
  float s = b_lin[c];
#pragma unroll 8
  for (int h = 0; h < HD; ++h) s += x[h] * w[h];
  out[idx] = s;   // out is [B,N,C] flat; rr == b*N+n
}

__global__ void zero_f32(float* p, int n) {
  int i = blockIdx.x * blockDim.x + threadIdx.x;
  if (i < n) p[i] = 0.f;
}

// ---------------------------------------------------------------------------
extern "C" void kernel_launch(void* const* d_in, const int* in_sizes, int n_in,
                              void* d_out, int out_size, void* d_ws, size_t ws_size,
                              hipStream_t stream) {
  const float* x     = (const float*)d_in[0];
  const int*   adj   = (const int*)  d_in[1];
  const float* Ws    = (const float*)d_in[2];
  const float* a_s   = (const float*)d_in[3];
  const float* W_out = (const float*)d_in[4];
  const float* a_out = (const float*)d_in[5];
  const float* W_ih  = (const float*)d_in[6];
  const float* W_hh  = (const float*)d_in[7];
  const float* b_ih  = (const float*)d_in[8];
  const float* b_hh  = (const float*)d_in[9];
  const float* W_lin = (const float*)d_in[10];
  const float* b_lin = (const float*)d_in[11];
  float* out = (float*)d_out;

  // workspace layout (floats); pre aliases Wh, x_seq aliases hcat
  float* ws   = (float*)d_ws;
  const long SZ_WH   = (long)KH * R1 * HD;        // 6,291,456
  const long SZ_HCAT = (long)R1 * KH * HD;        // 6,291,456
  const long SZ_WHO  = (long)R1 * HD;             // 1,572,864
  float* Wh   = ws;
  float* hcat = Wh + SZ_WH;
  float* WhO  = hcat + SZ_HCAT;
  float* f1   = WhO + SZ_WHO;                     // K*R1
  float* f2   = f1 + (long)KH * R1;
  float* f1o  = f2 + (long)KH * R1;
  float* f2o  = f1o + R1;
  float* gbuf = f2o + R1;                         // BN*256
  float* hh   = gbuf + (long)BN * 4 * HD;
  float* cc   = hh + (long)BN * HD;
  float* pre  = Wh;                               // alias: Wh dead after hcat built
  float* xseq = hcat;                             // alias: hcat dead after WhO/f's built

  // 1) Wh[k] = h @ Ws[k]   (24576x32 @ 32x64, 4 heads)
  gemm_f16wmma<<<dim3(R1 / 16, 1, KH), 128, 0, stream>>>(
      x, FF, Ws, HD, 1, FF, Wh, HD, HD, nullptr, nullptr,
      (long)FF * HD, (long)R1 * HD);
  // 2) f1/f2 per head
  f1f2_kernel<<<dim3(R1 / 256, 1, KH), 256, 0, stream>>>(Wh, a_s, f1, f2, (long)R1);
  // 3) head attention -> ELU -> concat layout hcat[M,N,256]
  attn_kernel<<<dim3(NN / 16, MM, KH), 128, 0, stream>>>(Wh, f1, f2, adj, hcat, 0);
  // 4) WhO = hcat @ W_out   (24576x256 @ 256x64)
  gemm_f16wmma<<<dim3(R1 / 16, 1, 1), 128, 0, stream>>>(
      hcat, KH * HD, W_out, HD, 1, KH * HD, WhO, HD, HD, nullptr, nullptr, 0, 0);
  // 5) f1o/f2o from a_out
  f1f2_kernel<<<dim3(R1 / 256, 1, 1), 256, 0, stream>>>(WhO, a_out, f1o, f2o, (long)R1);
  // 6) output-head attention -> ELU -> x_seq[BN,T,H]
  attn_kernel<<<dim3(NN / 16, MM, 1), 128, 0, stream>>>(WhO, f1o, f2o, adj, xseq, 1);
  // 7) pre = x_seq @ W_ih^T + b_ih + b_hh   (24576x64 @ 64x256), B strided-transposed
  gemm_f16wmma<<<dim3((BN * TT) / 16, 1, 1), 128, 0, stream>>>(
      xseq, HD, W_ih, 1, HD, HD, pre, 4 * HD, 4 * HD, b_ih, b_hh, 0, 0);
  // 8) h0 = c0 = 0 (hh, cc contiguous)
  zero_f32<<<(2 * BN * HD + 255) / 256, 256, 0, stream>>>(hh, 2 * BN * HD);
  // 9) LSTM scan over T steps
  for (int t = 0; t < TT; ++t) {
    gemm_f16wmma<<<dim3(BN / 16, 1, 1), 128, 0, stream>>>(
        hh, HD, W_hh, 1, HD, HD, gbuf, 4 * HD, 4 * HD, nullptr, nullptr, 0, 0);
    lstm_gate<<<(BN * HD) / 256, 256, 0, stream>>>(pre, gbuf, hh, cc, t);
  }
  // 10) out = hT @ W_lin^T + b_lin
  final_proj<<<(BN * CC_) / 256, 256, 0, stream>>>(hh, W_lin, b_lin, out);
}
